// EncoderDecoderModel_47691316855458
// MI455X (gfx1250) — matrix operand order repaired and checked by reading, not compiled, not run on hardware
//
#include <hip/hip_runtime.h>
#include <hip/hip_bf16.h>
#include <math.h>

#define B_   1024
#define S_   96
#define I_   128
#define H_   1024   // HE == HD
#define O_   128
#define HOR_ 24

typedef __attribute__((ext_vector_type(16))) __bf16 v16bf;
typedef __attribute__((ext_vector_type(8)))  float  v8f;

union Frag {
  v16bf  v;
  uint4  q[2];
  __bf16 e[16];
};

// A-matrix fragment (16x32 bf16, ISA 7.12.2): lane half=0 holds K {k0..k0+7, k0+16..k0+23},
// half=1 holds K {k0+8..k0+15, k0+24..k0+31}. Row-major source, K contiguous.
__device__ __forceinline__ v16bf ld_a_bf16(const __bf16* p, int half) {
  Frag u;
  u.q[0] = *(const uint4*)(p + half * 8);
  u.q[1] = *(const uint4*)(p + 16 + half * 8);
  return u.v;
}

// Same fragment but source is fp32 (convert inline to bf16).
__device__ __forceinline__ v16bf ld_a_f32(const float* p, int half) {
  float4 a = *(const float4*)(p + half * 8);
  float4 b = *(const float4*)(p + half * 8 + 4);
  float4 c = *(const float4*)(p + 16 + half * 8);
  float4 d = *(const float4*)(p + 16 + half * 8 + 4);
  Frag u;
  u.e[0]  = (__bf16)a.x; u.e[1]  = (__bf16)a.y; u.e[2]  = (__bf16)a.z; u.e[3]  = (__bf16)a.w;
  u.e[4]  = (__bf16)b.x; u.e[5]  = (__bf16)b.y; u.e[6]  = (__bf16)b.z; u.e[7]  = (__bf16)b.w;
  u.e[8]  = (__bf16)c.x; u.e[9]  = (__bf16)c.y; u.e[10] = (__bf16)c.z; u.e[11] = (__bf16)c.w;
  u.e[12] = (__bf16)d.x; u.e[13] = (__bf16)d.y; u.e[14] = (__bf16)d.z; u.e[15] = (__bf16)d.w;
  return u.v;
}

// B-matrix fragment (32x16 bf16): lane = column n (= row of row-major W), half=0 holds
// K {k0..k0+7, k0+16..k0+23}, half=1 holds K {k0+8..k0+15, k0+24..k0+31}.
__device__ __forceinline__ v16bf ld_b_bf16(const __bf16* p, int half) {
  Frag u;
  u.q[0] = *(const uint4*)(p + half * 8);
  u.q[1] = *(const uint4*)(p + 16 + half * 8);
  return u.v;
}

__device__ __forceinline__ v8f wmma_bf16(v16bf a, v16bf b, v8f c) {
  return __builtin_amdgcn_wmma_f32_16x16x32_bf16(false, a, false, b, (short)0, c,
                                                 false, false);
}

__device__ __forceinline__ float sigm(float x) { return 1.0f / (1.0f + __expf(-x)); }
__device__ __forceinline__ float tanh_f(float x) {
  x = fminf(fmaxf(x, -15.0f), 15.0f);
  float e = __expf(-2.0f * x);
  return (1.0f - e) / (1.0f + e);
}

// ---------------------------------------------------------------------------
// Fused LSTM cell step:  gates = x @ Wih^T + h @ Whh^T + bias ; update c, h.
// Block = 128 threads (4 waves). Per-wave tile: 32 rows x 32 gate-cols for all
// 4 gate blocks (each B fragment feeds 2 WMMAs -> ~1.25 loads/WMMA steady
// state). Block tile: 128 rows x 32 cols. Pointwise update fused in-register.
// x == nullptr  -> zero input (decoder step 0)
// h_in/c_in == nullptr -> zero initial state (encoder step 0)
// ---------------------------------------------------------------------------
__global__ __launch_bounds__(128, 1)
void lstm_step_kernel(const float*  __restrict__ x, long x_ld,
                      const __bf16* __restrict__ h_in,
                      const float*  __restrict__ c_in,
                      const __bf16* __restrict__ Wih,   // [4H, KI] bf16
                      const __bf16* __restrict__ Whh,   // [4H, H]  bf16
                      const float*  __restrict__ bias,  // [4H] = b_ih + b_hh
                      __bf16* __restrict__ h_out,       // [B, H] bf16
                      float*  __restrict__ c_out,       // [B, H] f32
                      int KI)
{
  const int lane = threadIdx.x & 31;
  const int wave = threadIdx.x >> 5;
  const int half = lane >> 4;
  const int r    = lane & 15;
  const int n0   = blockIdx.x * 32;               // gate-column tile
  const int m0   = blockIdx.y * 128 + wave * 32;  // batch-row tile (per wave)

  const v8f vzero = {0.f, 0.f, 0.f, 0.f, 0.f, 0.f, 0.f, 0.f};
  v8f acc[4][2][2];  // [gate][n-subtile][m-subtile]
  #pragma unroll
  for (int g = 0; g < 4; ++g)
    #pragma unroll
    for (int j = 0; j < 2; ++j)
      #pragma unroll
      for (int mi = 0; mi < 2; ++mi) acc[g][j][mi] = vzero;

  // ---- input contribution (fp32 source, converted inline) ----
  if (x) {
    const float* arow[2] = {x + (long)(m0 + r) * x_ld,
                            x + (long)(m0 + 16 + r) * x_ld};
    const __bf16* wrow[4][2];
    #pragma unroll
    for (int g = 0; g < 4; ++g)
      #pragma unroll
      for (int j = 0; j < 2; ++j)
        wrow[g][j] = Wih + (long)(g * H_ + n0 + j * 16 + r) * KI;
    for (int k0 = 0; k0 < KI; k0 += 32) {
      v16bf a0 = ld_a_f32(arow[0] + k0, half);
      v16bf a1 = ld_a_f32(arow[1] + k0, half);
      #pragma unroll
      for (int g = 0; g < 4; ++g) {
        #pragma unroll
        for (int j = 0; j < 2; ++j) {
          v16bf b = ld_b_bf16(wrow[g][j] + k0, half);
          acc[g][j][0] = wmma_bf16(a0, b, acc[g][j][0]);
          acc[g][j][1] = wmma_bf16(a1, b, acc[g][j][1]);
        }
      }
    }
  }

  // ---- recurrent contribution (bf16 h) ----
  if (h_in) {
    const __bf16* arow[2] = {h_in + (long)(m0 + r) * H_,
                             h_in + (long)(m0 + 16 + r) * H_};
    const __bf16* wrow[4][2];
    #pragma unroll
    for (int g = 0; g < 4; ++g)
      #pragma unroll
      for (int j = 0; j < 2; ++j)
        wrow[g][j] = Whh + (long)(g * H_ + n0 + j * 16 + r) * H_;
    for (int k0 = 0; k0 < H_; k0 += 32) {
      v16bf a0 = ld_a_bf16(arow[0] + k0, half);
      v16bf a1 = ld_a_bf16(arow[1] + k0, half);
      #pragma unroll
      for (int g = 0; g < 4; ++g) {
        #pragma unroll
        for (int j = 0; j < 2; ++j) {
          v16bf b = ld_b_bf16(wrow[g][j] + k0, half);
          acc[g][j][0] = wmma_bf16(a0, b, acc[g][j][0]);
          acc[g][j][1] = wmma_bf16(a1, b, acc[g][j][1]);
        }
      }
    }
  }

  // ---- fused pointwise LSTM cell update ----
  // C/D layout: element (VGPR v, lane) -> M = m0 + mi*16 + half*8 + v,
  //                                       N = n0 + j*16 + r
  #pragma unroll
  for (int j = 0; j < 2; ++j) {
    const int n    = n0 + j * 16 + r;
    const float bi = bias[n];
    const float bf = bias[H_ + n];
    const float bg = bias[2 * H_ + n];
    const float bo = bias[3 * H_ + n];
    #pragma unroll
    for (int mi = 0; mi < 2; ++mi) {
      #pragma unroll
      for (int v = 0; v < 8; ++v) {
        const long m    = m0 + mi * 16 + half * 8 + v;
        const float co  = c_in ? c_in[m * H_ + n] : 0.0f;
        const float ig  = sigm(acc[0][j][mi][v] + bi);
        const float fg  = sigm(acc[1][j][mi][v] + bf);
        const float gg  = tanh_f(acc[2][j][mi][v] + bg);
        const float og  = sigm(acc[3][j][mi][v] + bo);
        const float cn  = fg * co + ig * gg;
        c_out[m * H_ + n] = cn;
        h_out[m * H_ + n] = (__bf16)(og * tanh_f(cn));
      }
    }
  }
}

// ---------------------------------------------------------------------------
// Generic linear:  Y = X @ W^T + bias   (X bf16 [M,K], W bf16 [N,K])
// Per-wave tile 32x32; block tile 128x32.
// Optional fp32 output (strided) and/or bf16 output. Used for bridge + out-proj.
// ---------------------------------------------------------------------------
__global__ __launch_bounds__(128, 1)
void linear_kernel(const __bf16* __restrict__ X,
                   const __bf16* __restrict__ W,
                   const float*  __restrict__ bias,
                   float*  __restrict__ yf, long yf_ld,
                   __bf16* __restrict__ yb, long yb_ld,
                   int K)
{
  const int lane = threadIdx.x & 31;
  const int wave = threadIdx.x >> 5;
  const int half = lane >> 4;
  const int r    = lane & 15;
  const int n0   = blockIdx.x * 32;
  const int m0   = blockIdx.y * 128 + wave * 32;

  const v8f vzero = {0.f, 0.f, 0.f, 0.f, 0.f, 0.f, 0.f, 0.f};
  v8f acc[2][2] = {{vzero, vzero}, {vzero, vzero}};  // [n-subtile][m-subtile]

  const __bf16* arow[2] = {X + (long)(m0 + r) * K, X + (long)(m0 + 16 + r) * K};
  const __bf16* wrow[2] = {W + (long)(n0 + r) * K, W + (long)(n0 + 16 + r) * K};

  for (int k0 = 0; k0 < K; k0 += 32) {
    v16bf a0 = ld_a_bf16(arow[0] + k0, half);
    v16bf a1 = ld_a_bf16(arow[1] + k0, half);
    #pragma unroll
    for (int j = 0; j < 2; ++j) {
      v16bf b = ld_b_bf16(wrow[j] + k0, half);
      acc[j][0] = wmma_bf16(a0, b, acc[j][0]);
      acc[j][1] = wmma_bf16(a1, b, acc[j][1]);
    }
  }

  #pragma unroll
  for (int j = 0; j < 2; ++j) {
    const int n    = n0 + j * 16 + r;
    const float bb = bias[n];
    #pragma unroll
    for (int mi = 0; mi < 2; ++mi) {
      #pragma unroll
      for (int v = 0; v < 8; ++v) {
        const long m  = m0 + mi * 16 + half * 8 + v;
        const float y = acc[j][mi][v] + bb;
        if (yf) yf[m * yf_ld + n] = y;
        if (yb) yb[m * yb_ld + n] = (__bf16)y;
      }
    }
  }
}

__global__ void f32_to_bf16_kernel(const float* __restrict__ src,
                                   __bf16* __restrict__ dst, int n) {
  int i = blockIdx.x * 256 + threadIdx.x;
  if (i < n) dst[i] = (__bf16)src[i];
}

__global__ void vec_add_kernel(const float* __restrict__ a,
                               const float* __restrict__ b,
                               float* __restrict__ dst, int n) {
  int i = blockIdx.x * 256 + threadIdx.x;
  if (i < n) dst[i] = a[i] + b[i];
}

// ---------------------------------------------------------------------------
extern "C" void kernel_launch(void* const* d_in, const int* in_sizes, int n_in,
                              void* d_out, int out_size, void* d_ws, size_t ws_size,
                              hipStream_t stream) {
  (void)in_sizes; (void)n_in; (void)out_size; (void)ws_size;
  const float* x     = (const float*)d_in[0];
  const float* Wih_e = (const float*)d_in[1];
  const float* Whh_e = (const float*)d_in[2];
  const float* bih_e = (const float*)d_in[3];
  const float* bhh_e = (const float*)d_in[4];
  const float* Wih_d = (const float*)d_in[5];
  const float* Whh_d = (const float*)d_in[6];
  const float* bih_d = (const float*)d_in[7];
  const float* bhh_d = (const float*)d_in[8];
  const float* Wfc   = (const float*)d_in[9];
  const float* bfc   = (const float*)d_in[10];
  const float* Wfcc  = (const float*)d_in[11];
  const float* bfcc  = (const float*)d_in[12];
  const float* Wout  = (const float*)d_in[13];
  const float* bout  = (const float*)d_in[14];
  float* out = (float*)d_out;

  // ---- carve workspace (~35 MB) ----
  char* p = (char*)d_ws;
  auto carve = [&](size_t bytes) -> void* {
    void* r = (void*)p;
    p += (bytes + 255) & ~(size_t)255;
    return r;
  };
  __bf16* Wih_e_b = (__bf16*)carve((size_t)4 * H_ * I_ * 2);
  __bf16* Whh_e_b = (__bf16*)carve((size_t)4 * H_ * H_ * 2);
  __bf16* Wih_d_b = (__bf16*)carve((size_t)4 * H_ * O_ * 2);
  __bf16* Whh_d_b = (__bf16*)carve((size_t)4 * H_ * H_ * 2);
  __bf16* Wfc_b   = (__bf16*)carve((size_t)H_ * H_ * 2);
  __bf16* Wfcc_b  = (__bf16*)carve((size_t)H_ * H_ * 2);
  __bf16* Wout_b  = (__bf16*)carve((size_t)O_ * H_ * 2);
  float*  bias_e  = (float*)carve((size_t)4 * H_ * 4);
  float*  bias_d  = (float*)carve((size_t)4 * H_ * 4);
  __bf16* hbuf[2] = {(__bf16*)carve((size_t)B_ * H_ * 2),
                     (__bf16*)carve((size_t)B_ * H_ * 2)};
  float*  cbuf[2] = {(float*)carve((size_t)B_ * H_ * 4),
                     (float*)carve((size_t)B_ * H_ * 4)};

  auto cvt = [&](const float* s, __bf16* d, int n) {
    f32_to_bf16_kernel<<<(n + 255) / 256, 256, 0, stream>>>(s, d, n);
  };
  cvt(Wih_e, Wih_e_b, 4 * H_ * I_);
  cvt(Whh_e, Whh_e_b, 4 * H_ * H_);
  cvt(Wih_d, Wih_d_b, 4 * H_ * O_);
  cvt(Whh_d, Whh_d_b, 4 * H_ * H_);
  cvt(Wfc,   Wfc_b,   H_ * H_);
  cvt(Wfcc,  Wfcc_b,  H_ * H_);
  cvt(Wout,  Wout_b,  O_ * H_);
  vec_add_kernel<<<(4 * H_ + 255) / 256, 256, 0, stream>>>(bih_e, bhh_e, bias_e, 4 * H_);
  vec_add_kernel<<<(4 * H_ + 255) / 256, 256, 0, stream>>>(bih_d, bhh_d, bias_d, 4 * H_);

  const dim3 blk(128);
  const dim3 grd_lstm(H_ / 32, B_ / 128);  // 32 x 8 blocks
  const dim3 grd_out(O_ / 32, B_ / 128);   // 4 x 8 blocks

  // ---- encoder: 96 sequential steps (ping-pong state) ----
  int cur = 0, nxt = 1;
  for (int t = 0; t < S_; ++t) {
    lstm_step_kernel<<<grd_lstm, blk, 0, stream>>>(
        x + (long)t * I_, (long)S_ * I_,
        (t == 0) ? nullptr : hbuf[cur],
        (t == 0) ? nullptr : cbuf[cur],
        Wih_e_b, Whh_e_b, bias_e, hbuf[nxt], cbuf[nxt], I_);
    cur ^= 1; nxt ^= 1;
  }

  // ---- bridge: h_dec = h_enc @ Wfc^T + bfc (bf16), c_dec = h_enc @ Wfcc^T + bfcc (f32)
  linear_kernel<<<grd_lstm, blk, 0, stream>>>(hbuf[cur], Wfc_b, bfc,
                                              nullptr, 0, hbuf[nxt], (long)H_, H_);
  linear_kernel<<<grd_lstm, blk, 0, stream>>>(hbuf[cur], Wfcc_b, bfcc,
                                              cbuf[nxt], (long)H_, nullptr, 0, H_);
  cur ^= 1; nxt ^= 1;

  // ---- decoder: 24 autoregressive steps, feedback read from d_out ----
  for (int t = 0; t < HOR_; ++t) {
    lstm_step_kernel<<<grd_lstm, blk, 0, stream>>>(
        (t == 0) ? nullptr : (out + (long)(t - 1) * O_), (long)HOR_ * O_,
        hbuf[cur], cbuf[cur],
        Wih_d_b, Whh_d_b, bias_d, hbuf[nxt], cbuf[nxt], O_);
    cur ^= 1; nxt ^= 1;
    linear_kernel<<<grd_out, blk, 0, stream>>>(hbuf[cur], Wout_b, bout,
                                               out + (long)t * O_, (long)HOR_ * O_,
                                               nullptr, 0, H_);
  }
}
